// HybridAttnResidualTransformerBlock_51041391346393
// MI455X (gfx1250) — matrix operand order repaired and checked by reading, not compiled, not run on hardware
//
#include <hip/hip_runtime.h>

typedef __attribute__((ext_vector_type(16))) __bf16 bf16x16;
typedef __attribute__((ext_vector_type(8)))  __bf16 bf16x8;
typedef __attribute__((ext_vector_type(8)))  float  f32x8;
typedef __attribute__((ext_vector_type(4)))  unsigned u32x4;
typedef __attribute__((ext_vector_type(8)))  int      i32x8;
typedef __attribute__((ext_vector_type(4)))  int      i32x4;

union BfFrag { bf16x16 v; bf16x8 h[2]; };

__device__ __forceinline__ __bf16 f2bf(float f) {
  union { float f; unsigned u; } a; a.f = f;
  unsigned u = a.u;
  unsigned r = (u + 0x7FFFu + ((u >> 16) & 1u)) >> 16;  // round-to-nearest-even
  union { unsigned short s; __bf16 b; } o; o.s = (unsigned short)r;
  return o.b;
}

__device__ __forceinline__ float gelu_tanh(float x) {
  float x3 = x * x * x;
  float z  = 0.7978845608028654f * (x + 0.044715f * x3);
  float e  = __expf(2.0f * z);
  float th = 1.0f - 2.0f / (e + 1.0f);   // tanh(z), overflow-safe
  return 0.5f * x * (1.0f + th);
}

__device__ __forceinline__ unsigned lds_offset(const void* p) {
  // Flat LDS aperture: addr[31:0] is the wave-relative LDS byte offset.
  return (unsigned)(unsigned long long)p;
}

// ---------------------------------------------------------------------------
// TDM: DMA a 2D bf16 tile (tileK x tileRows) from global [rows, K] row-major
// into LDS (row-major tile, pitch tileK). D# per CDNA5 ISA ch.8.
// clang-23 toolchain: 6-arg builtin (g0, g1, g2, g3, extra, cpol).
// ---------------------------------------------------------------------------
__device__ __forceinline__ void tdm_load_tile_bf16(
    const void* gtile, unsigned ldsOff, unsigned tensorK, unsigned tensorRows,
    unsigned tileK, unsigned tileRows)
{
  unsigned long long ga = (unsigned long long)gtile;
  u32x4 g0;
  g0[0] = 1u;                                            // count=1, user mode
  g0[1] = ldsOff;                                        // lds_addr [63:32]
  g0[2] = (unsigned)(ga & 0xFFFFFFFFull);                // global_addr[31:0]
  g0[3] = (unsigned)((ga >> 32) & 0x01FFFFFFull)         // global_addr[56:32]
        | (2u << 30);                                    // type=2 (image)
  i32x8 g1;
  g1[0] = (int)(1u << 16);                               // data_size=1 (2 bytes)
  g1[1] = (int)((tensorK & 0xFFFFu) << 16);              // tensor_dim0[15:0] @63:48
  g1[2] = (int)((tensorK >> 16) | ((tensorRows & 0xFFFFu) << 16)); // dim0 hi | dim1 lo
  g1[3] = (int)((tensorRows >> 16) | (tileK << 16));     // dim1 hi | tile_dim0 @127:112
  g1[4] = (int)tileRows;                                 // tile_dim1 @143:128 (tile_dim2=0)
  g1[5] = (int)tensorK;                                  // tensor_dim0_stride[31:0] (row pitch)
  g1[6] = 0;                                             // stride0 hi | stride1 lo
  g1[7] = 0;
  i32x4 z4; z4[0] = 0; z4[1] = 0; z4[2] = 0; z4[3] = 0;  // groups 2/3: 2D tensor
  i32x8 z8;
  #pragma unroll
  for (int i = 0; i < 8; ++i) z8[i] = 0;
  __builtin_amdgcn_tensor_load_to_lds(g0, g1, z4, z4, z8, 0);
}

// ---------------------------------------------------------------------------
// Weight fp32 [K,N] -> bf16 transposed [N,K]
// ---------------------------------------------------------------------------
__global__ __launch_bounds__(256) void transpose_bf16_kernel(
    const float* __restrict__ W, __bf16* __restrict__ Wt, int K, int N)
{
  long idx = (long)blockIdx.x * 256 + threadIdx.x;
  if (idx < (long)K * N) {
    int n = (int)(idx / K);
    int k = (int)(idx % K);
    Wt[idx] = f2bf(W[(size_t)k * N + n]);
  }
}

// ---------------------------------------------------------------------------
// Block-wide reduction of a pair of floats (256 threads = 8 waves of 32)
// ---------------------------------------------------------------------------
__device__ __forceinline__ void block_reduce2(float& a, float& b) {
  #pragma unroll
  for (int off = 16; off > 0; off >>= 1) {
    a += __shfl_xor(a, off, 32);
    b += __shfl_xor(b, off, 32);
  }
  __shared__ float ra[8], rb[8];
  int wid = threadIdx.x >> 5, lane = threadIdx.x & 31;
  __syncthreads();
  if (lane == 0) { ra[wid] = a; rb[wid] = b; }
  __syncthreads();
  float sa = 0.f, sb = 0.f;
  #pragma unroll
  for (int i = 0; i < 8; ++i) { sa += ra[i]; sb += rb[i]; }
  a = sa; b = sb;
}

// ---------------------------------------------------------------------------
// Fused block_attn_res + layer_norm. One block per token (B*T blocks).
// ---------------------------------------------------------------------------
__global__ __launch_bounds__(256) void block_res_ln_kernel(
    const float* __restrict__ blocks, const float* __restrict__ extra,
    const float* __restrict__ rw, const float* __restrict__ rg,
    const float* __restrict__ lg, const float* __restrict__ lb,
    __bf16* __restrict__ out, int D, int NB, size_t blockStride)
{
  const int token = blockIdx.x;
  const int tid   = threadIdx.x;
  const int E     = D >> 8;                 // elements per thread (4 for D=1024)
  const float invD = 1.0f / (float)D;

  float v[5][4];
  float pwg[4];
  float logit[5];

  #pragma unroll
  for (int e = 0; e < 4; ++e) if (e < E) {
    int d = e * 256 + tid;
    pwg[e] = rw[d] * rg[d];
  }

  for (int n = 0; n <= NB; ++n) {
    const float* src = (n < NB) ? (blocks + (size_t)n * blockStride + (size_t)token * D)
                                : (extra + (size_t)token * D);
    float s2 = 0.f, sd = 0.f;
    #pragma unroll
    for (int e = 0; e < 4; ++e) if (e < E) {
      int d = e * 256 + tid;
      float x = src[d];
      v[n][e] = x;
      s2 += x * x;
      sd += pwg[e] * x;
    }
    block_reduce2(s2, sd);
    logit[n] = sd * rsqrtf(s2 * invD + 1e-6f);   // rms_norm folded into the dot
  }

  float mx = logit[0];
  for (int n = 1; n <= NB; ++n) mx = fmaxf(mx, logit[n]);
  float w[5], wsum = 0.f;
  for (int n = 0; n <= NB; ++n) { w[n] = __expf(logit[n] - mx); wsum += w[n]; }
  float winv = 1.0f / wsum;

  float h[4], s = 0.f, s2 = 0.f;
  #pragma unroll
  for (int e = 0; e < 4; ++e) if (e < E) {
    float acc = 0.f;
    for (int n = 0; n <= NB; ++n) acc += w[n] * v[n][e];
    acc *= winv;
    h[e] = acc;
    s  += acc;
    s2 += acc * acc;
  }
  block_reduce2(s, s2);
  float mean = s * invD;
  float var  = s2 * invD - mean * mean;
  float rstd = rsqrtf(var + 1e-5f);

  #pragma unroll
  for (int e = 0; e < 4; ++e) if (e < E) {
    int d = e * 256 + tid;
    out[(size_t)token * D + d] = f2bf((h[e] - mean) * rstd * lg[d] + lb[d]);
  }
}

// ---------------------------------------------------------------------------
// bf16 GEMM with TDM double-buffered LDS staging.
// C[M,N] = A[M,K] (row-major bf16) x Bt[N,K] (bf16, B pre-transposed).
// Block = 256 threads = 8 waves; block tile 128x128; wave tile 32x64.
// Per K-step: wave0 issues TENSOR_LOAD_TO_LDS for next A/B tiles (128x32 bf16
// each), all waves compute 8 WMMAs from current LDS buffers.
// EPI: 0 = f32 store, 2 = gelu -> bf16 store, 3 = +Res -> f32 store.
// ---------------------------------------------------------------------------
template<int EPI>
__global__ __launch_bounds__(256) void gemm_bf16_kernel(
    const __bf16* __restrict__ A, const __bf16* __restrict__ Bt,
    int M, int N, int K,
    const float* __restrict__ Res, float* __restrict__ Cf, __bf16* __restrict__ Cb)
{
  __shared__ __attribute__((aligned(128))) __bf16 sA[2][128 * 32];
  __shared__ __attribute__((aligned(128))) __bf16 sB[2][128 * 32];

  const int lane  = threadIdx.x & 31, wid = threadIdx.x >> 5;
  const int lm    = lane & 15, lh = lane >> 4;
  const int waveM = wid & 3, waveN = wid >> 2;          // 4 x 2 wave grid
  const int mBase = blockIdx.y * 128;
  const int nBase = blockIdx.x * 128;

  f32x8 zero = {};
  f32x8 acc[2][4];
  #pragma unroll
  for (int mt = 0; mt < 2; ++mt)
    #pragma unroll
    for (int nt = 0; nt < 4; ++nt) acc[mt][nt] = zero;

  const __bf16* Ablk = A  + (size_t)mBase * K;
  const __bf16* Bblk = Bt + (size_t)nBase * K;
  unsigned ldsA[2] = { lds_offset(&sA[0][0]), lds_offset(&sA[1][0]) };
  unsigned ldsB[2] = { lds_offset(&sB[0][0]), lds_offset(&sB[1][0]) };

  if (wid == 0) {
    tdm_load_tile_bf16(Ablk, ldsA[0], (unsigned)K, (unsigned)(M - mBase), 32u, 128u);
    tdm_load_tile_bf16(Bblk, ldsB[0], (unsigned)K, (unsigned)(N - nBase), 32u, 128u);
    __builtin_amdgcn_s_wait_tensorcnt(0);
  }
  __syncthreads();

  int buf = 0;
  for (int k0 = 0; k0 < K; k0 += 32, buf ^= 1) {
    if (wid == 0 && k0 + 32 < K) {
      tdm_load_tile_bf16(Ablk + k0 + 32, ldsA[buf ^ 1], (unsigned)K, (unsigned)(M - mBase), 32u, 128u);
      tdm_load_tile_bf16(Bblk + k0 + 32, ldsB[buf ^ 1], (unsigned)K, (unsigned)(N - nBase), 32u, 128u);
    }

    // A fragments (16x32): lane lm = row; K = 16*lh+{0..15}.
    BfFrag af[2];
    #pragma unroll
    for (int mt = 0; mt < 2; ++mt) {
      const __bf16* p = &sA[buf][(waveM * 32 + mt * 16 + lm) * 32 + 16 * lh];
      af[mt].h[0] = *(const bf16x8*)(p);
      af[mt].h[1] = *(const bf16x8*)(p + 8);
    }
    // B fragments (32x16): lane lm = column (row of Bt tile); K = 16*lh+{0..15}.
    BfFrag bf[4];
    #pragma unroll
    for (int nt = 0; nt < 4; ++nt) {
      const __bf16* p = &sB[buf][(waveN * 64 + nt * 16 + lm) * 32 + 16 * lh];
      bf[nt].h[0] = *(const bf16x8*)(p);
      bf[nt].h[1] = *(const bf16x8*)(p + 8);
    }
    #pragma unroll
    for (int mt = 0; mt < 2; ++mt)
      #pragma unroll
      for (int nt = 0; nt < 4; ++nt)
        acc[mt][nt] = __builtin_amdgcn_wmma_f32_16x16x32_bf16(
            false, af[mt].v, false, bf[nt].v, (short)0, acc[mt][nt], false, false);

    if (wid == 0) __builtin_amdgcn_s_wait_tensorcnt(0);
    __syncthreads();
  }

  #pragma unroll
  for (int mt = 0; mt < 2; ++mt)
  #pragma unroll
  for (int nt = 0; nt < 4; ++nt)
  #pragma unroll
  for (int r = 0; r < 8; ++r) {
    int row = mBase + waveM * 32 + mt * 16 + r + 8 * lh;   // C layout
    int col = nBase + waveN * 64 + nt * 16 + lm;
    size_t idx = (size_t)row * N + col;
    float val = acc[mt][nt][r];
    if (EPI == 0)      Cf[idx] = val;
    else if (EPI == 2) Cb[idx] = f2bf(gelu_tanh(val));
    else if (EPI == 3) Cf[idx] = val + Res[idx];
  }
}

// ---------------------------------------------------------------------------
// RoPE + pack: qkv fp32 [BT, 3D] -> Q,K bf16 [B,H,T,64] (Q pre-scaled 1/8),
// Vt bf16 [B,H,64,T]. One thread per rotation pair.
// ---------------------------------------------------------------------------
__global__ __launch_bounds__(256) void rope_pack_kernel(
    const float* __restrict__ qkv, const float* __restrict__ cosT,
    const float* __restrict__ sinT, __bf16* __restrict__ qb,
    __bf16* __restrict__ kb, __bf16* __restrict__ vbt,
    int T, int H, int D, int total)
{
  int idx = blockIdx.x * 256 + threadIdx.x;
  if (idx >= total) return;
  int per   = H * 32;
  int token = idx / per;
  int r     = idx % per;
  int h     = r >> 5;
  int i     = r & 31;
  int t     = token % T;
  int b     = token / T;

  const float* p = qkv + (size_t)token * 3 * D;
  float c1 = cosT[t * 64 + i],      s1 = sinT[t * 64 + i];
  float c2 = cosT[t * 64 + i + 32], s2 = sinT[t * 64 + i + 32];
  int col = h * 64 + i;
  float q1 = p[col],         q2 = p[col + 32];
  float k1 = p[D + col],     k2 = p[D + col + 32];
  float v1 = p[2 * D + col], v2 = p[2 * D + col + 32];

  size_t hb = ((size_t)(b * H + h) * T + t) * 64;
  qb[hb + i]      = f2bf((q1 * c1 - q2 * s1) * 0.125f);   // 1/sqrt(64) folded in
  qb[hb + i + 32] = f2bf((q2 * c2 + q1 * s2) * 0.125f);
  kb[hb + i]      = f2bf(k1 * c1 - k2 * s1);
  kb[hb + i + 32] = f2bf(k2 * c2 + k1 * s2);

  size_t vb = ((size_t)(b * H + h) * 64 + i) * T + t;
  vbt[vb]                  = f2bf(v1);
  vbt[vb + (size_t)32 * T] = f2bf(v2);
}

// ---------------------------------------------------------------------------
// Flash attention (causal): one wave per 16-query tile per (b,h).
// Q,K: [B,H,T,64] bf16; Vt: [B,H,64,T] bf16; O: [BT, D] bf16.
// ---------------------------------------------------------------------------
__global__ __launch_bounds__(256) void attn_kernel(
    const __bf16* __restrict__ Q, const __bf16* __restrict__ Kc,
    const __bf16* __restrict__ VT, __bf16* __restrict__ O,
    int T, int H, int D, int totalWaves)
{
  const int lane = threadIdx.x & 31;
  const int gw   = blockIdx.x * 8 + (threadIdx.x >> 5);
  if (gw >= totalWaves) return;
  const int qtiles = T / 16;
  const int qt = gw % qtiles;
  const int h  = (gw / qtiles) % H;
  const int b  = gw / (qtiles * H);
  const int q0 = qt * 16;
  const int lm = lane & 15, lh = lane >> 4;
  const float NEG = -3.0e38f;

  const size_t head = (size_t)(b * H + h) * T * 64;

  // Q fragments (16x64 = two 16x32 A-fragments), loaded once.
  BfFrag qf[2];
  const __bf16* qp = Q + head + (size_t)(q0 + lm) * 64 + 16 * lh;
  qf[0].h[0] = *(const bf16x8*)(qp);      qf[0].h[1] = *(const bf16x8*)(qp + 8);
  qf[1].h[0] = *(const bf16x8*)(qp + 32); qf[1].h[1] = *(const bf16x8*)(qp + 40);

  f32x8 zero = {};
  f32x8 oacc[4]; oacc[0] = zero; oacc[1] = zero; oacc[2] = zero; oacc[3] = zero;
  float mrow[8], lrow[8];
  #pragma unroll
  for (int r = 0; r < 8; ++r) { mrow[r] = NEG; lrow[r] = 0.f; }

  __shared__ __attribute__((aligned(16))) __bf16 lp[8][16 * 32];
  __bf16* myp = lp[threadIdx.x >> 5];

  const int kbMax = (q0 + 15) / 32;
  for (int kbi = 0; kbi <= kbMax; ++kbi) {
    const int kbase = kbi * 32;

    // S = Q x K^T for two 16-key tiles (K-dim 64 -> 2 WMMA each)
    f32x8 s[2];
    #pragma unroll
    for (int j = 0; j < 2; ++j) {
      const __bf16* kp = Kc + head + (size_t)(kbase + j * 16 + lm) * 64 + 16 * lh;
      BfFrag kf0, kf1;
      kf0.h[0] = *(const bf16x8*)(kp);      kf0.h[1] = *(const bf16x8*)(kp + 8);
      kf1.h[0] = *(const bf16x8*)(kp + 32); kf1.h[1] = *(const bf16x8*)(kp + 40);
      f32x8 z = zero;
      z = __builtin_amdgcn_wmma_f32_16x16x32_bf16(false, qf[0].v, false, kf0.v, (short)0, z, false, false);
      z = __builtin_amdgcn_wmma_f32_16x16x32_bf16(false, qf[1].v, false, kf1.v, (short)0, z, false, false);
      s[j] = z;
    }

    // Causal mask only needed on the diagonal chunk (wave-uniform branch).
    if (kbase + 31 > q0) {
      #pragma unroll
      for (int r = 0; r < 8; ++r) {
        int qrow = q0 + r + 8 * lh;
        s[0][r] = (kbase + lm      <= qrow) ? s[0][r] : NEG;
        s[1][r] = (kbase + 16 + lm <= qrow) ? s[1][r] : NEG;
      }
    }

    // Online softmax update (rows live in 16-lane halves)
    #pragma unroll
    for (int r = 0; r < 8; ++r) {
      float v0 = s[0][r], v1 = s[1][r];
      float mx = fmaxf(v0, v1);
      mx = fmaxf(mx, __shfl_xor(mx, 1, 32));
      mx = fmaxf(mx, __shfl_xor(mx, 2, 32));
      mx = fmaxf(mx, __shfl_xor(mx, 4, 32));
      mx = fmaxf(mx, __shfl_xor(mx, 8, 32));
      float mnew = fmaxf(mrow[r], mx);
      float f  = __expf(mrow[r] - mnew);
      float p0 = __expf(v0 - mnew);
      float p1 = __expf(v1 - mnew);
      float rs = p0 + p1;
      rs += __shfl_xor(rs, 1, 32);
      rs += __shfl_xor(rs, 2, 32);
      rs += __shfl_xor(rs, 4, 32);
      rs += __shfl_xor(rs, 8, 32);
      lrow[r] = lrow[r] * f + rs;
      mrow[r] = mnew;
      #pragma unroll
      for (int t4 = 0; t4 < 4; ++t4) oacc[t4][r] *= f;
      // Stage P (C layout) into LDS as bf16 for re-fragmentation
      myp[(r + 8 * lh) * 32 + lm]      = f2bf(p0);
      myp[(r + 8 * lh) * 32 + 16 + lm] = f2bf(p1);
    }
    asm volatile("s_wait_dscnt 0x0" ::: "memory");

    // Reload P as a 16x32 A-fragment
    BfFrag pa;
    pa.h[0] = *(const bf16x8*)(&myp[lm * 32 + 16 * lh]);
    pa.h[1] = *(const bf16x8*)(&myp[lm * 32 + 16 * lh + 8]);
    asm volatile("s_wait_dscnt 0x0" ::: "memory");

    // O += P x V   (V^T rows are contiguous per hd index)
    #pragma unroll
    for (int t4 = 0; t4 < 4; ++t4) {
      const __bf16* vp = VT + ((size_t)(b * H + h) * 64 + t4 * 16 + lm) * T + kbase + 16 * lh;
      BfFrag vf;
      vf.h[0] = *(const bf16x8*)(vp);
      vf.h[1] = *(const bf16x8*)(vp + 8);
      oacc[t4] = __builtin_amdgcn_wmma_f32_16x16x32_bf16(false, pa.v, false, vf.v, (short)0, oacc[t4], false, false);
    }
  }

  // Normalize and store as [B,T,D] bf16 (D col = h*64 + hd)
  #pragma unroll
  for (int r = 0; r < 8; ++r) {
    float inv = 1.0f / lrow[r];
    int q = q0 + r + 8 * lh;
    size_t rowBase = (size_t)(b * T + q) * D + h * 64;
    #pragma unroll
    for (int t4 = 0; t4 < 4; ++t4)
      O[rowBase + t4 * 16 + lm] = f2bf(oacc[t4][r] * inv);
  }
}

// ---------------------------------------------------------------------------
// Host orchestration
// ---------------------------------------------------------------------------
extern "C" void kernel_launch(void* const* d_in, const int* in_sizes, int n_in,
                              void* d_out, int out_size, void* d_ws, size_t ws_size,
                              hipStream_t stream)
{
  const float* blocks     = (const float*)d_in[0];
  const float* hidden     = (const float*)d_in[1];
  const float* ln1_g      = (const float*)d_in[2];
  const float* ln1_b      = (const float*)d_in[3];
  const float* ln2_g      = (const float*)d_in[4];
  const float* ln2_b      = (const float*)d_in[5];
  const float* attn_res_w = (const float*)d_in[6];
  const float* attn_res_g = (const float*)d_in[7];
  const float* mlp_res_w  = (const float*)d_in[8];
  const float* mlp_res_g  = (const float*)d_in[9];
  const float* Wq         = (const float*)d_in[10];
  const float* Wk         = (const float*)d_in[11];
  const float* Wv         = (const float*)d_in[12];
  const float* Wo         = (const float*)d_in[13];
  const float* W1         = (const float*)d_in[14];
  const float* W2         = (const float*)d_in[15];
  const float* rope_cos   = (const float*)d_in[16];
  const float* rope_sin   = (const float*)d_in[17];

  const int D  = in_sizes[2];
  const int T  = in_sizes[16] / 64;
  const int BT = in_sizes[1] / D;
  const int B  = BT / T;
  const int H  = D / 64;
  const int F  = in_sizes[14] / D;
  const int NB = in_sizes[0] / in_sizes[1];

  size_t off = 0;
  auto carve = [&](size_t bytes) -> void* {
    void* p = (char*)d_ws + off;
    off += (bytes + 255) & ~(size_t)255;
    return p;
  };
  __bf16* wqkv_t  = (__bf16*)carve((size_t)3 * D * D * 2);
  __bf16* wo_t    = (__bf16*)carve((size_t)D * D * 2);
  __bf16* w1_t    = (__bf16*)carve((size_t)F * D * 2);
  __bf16* w2_t    = (__bf16*)carve((size_t)D * F * 2);
  __bf16* xln     = (__bf16*)carve((size_t)BT * D * 2);
  __bf16* qb      = (__bf16*)carve((size_t)BT * D * 2);
  __bf16* kb      = (__bf16*)carve((size_t)BT * D * 2);
  __bf16* vbt     = (__bf16*)carve((size_t)BT * D * 2);
  __bf16* ob      = (__bf16*)carve((size_t)BT * D * 2);
  float*  partial = (float*)carve((size_t)BT * D * 4);
  size_t bigBytes = (size_t)BT * 3 * D * 4;            // qkv fp32
  size_t ffBytes  = (size_t)BT * F * 2;                // ff1 bf16 (used later)
  if (ffBytes > bigBytes) bigBytes = ffBytes;
  void* big = carve(bigBytes);                         // qkv and ff1 alias (disjoint lifetimes)
  float*  qkv = (float*)big;
  __bf16* ff1 = (__bf16*)big;
  (void)ws_size; (void)n_in; (void)out_size;

  auto tgrid = [](long n) { return (unsigned)((n + 255) / 256); };

  // Weight conversion: fp32 [K,N] -> bf16 [N,K]
  transpose_bf16_kernel<<<tgrid((long)D * D), 256, 0, stream>>>(Wq, wqkv_t, D, D);
  transpose_bf16_kernel<<<tgrid((long)D * D), 256, 0, stream>>>(Wk, wqkv_t + (size_t)D * D, D, D);
  transpose_bf16_kernel<<<tgrid((long)D * D), 256, 0, stream>>>(Wv, wqkv_t + (size_t)2 * D * D, D, D);
  transpose_bf16_kernel<<<tgrid((long)D * D), 256, 0, stream>>>(Wo, wo_t, D, D);
  transpose_bf16_kernel<<<tgrid((long)D * F), 256, 0, stream>>>(W1, w1_t, D, F);
  transpose_bf16_kernel<<<tgrid((long)F * D), 256, 0, stream>>>(W2, w2_t, F, D);

  // 1. block_attn_res + LN1 -> xln (bf16)
  block_res_ln_kernel<<<BT, 256, 0, stream>>>(blocks, hidden, attn_res_w, attn_res_g,
                                              ln1_g, ln1_b, xln, D, NB, (size_t)BT * D);
  // 2. QKV projection (fp32 out)
  gemm_bf16_kernel<0><<<dim3(3 * D / 128, BT / 128), 256, 0, stream>>>(
      xln, wqkv_t, BT, 3 * D, D, nullptr, qkv, nullptr);
  // 3. RoPE + pack to bf16 (Q scaled by 1/sqrt(hd))
  {
    long total = (long)BT * H * 32;
    rope_pack_kernel<<<tgrid(total), 256, 0, stream>>>(qkv, rope_cos, rope_sin,
                                                       qb, kb, vbt, T, H, D, (int)total);
  }
  // 4. Causal flash attention -> ob (bf16 [BT, D])
  {
    int waves = B * H * (T / 16);
    attn_kernel<<<(waves + 7) / 8, 256, 0, stream>>>(qb, kb, vbt, ob, T, H, D, waves);
  }
  // 5. Output projection + residual(hidden) -> partial (fp32)
  gemm_bf16_kernel<3><<<dim3(D / 128, BT / 128), 256, 0, stream>>>(
      ob, wo_t, BT, D, D, hidden, partial, nullptr);
  // 6. block_attn_res + LN2 -> xln (bf16)
  block_res_ln_kernel<<<BT, 256, 0, stream>>>(blocks, partial, mlp_res_w, mlp_res_g,
                                              ln2_g, ln2_b, xln, D, NB, (size_t)BT * D);
  // 7. FF1 + gelu -> ff1 (bf16)
  gemm_bf16_kernel<2><<<dim3(F / 128, BT / 128), 256, 0, stream>>>(
      xln, w1_t, BT, F, D, nullptr, nullptr, ff1);
  // 8. FF2 + residual(partial) -> d_out (fp32)
  gemm_bf16_kernel<3><<<dim3(D / 128, BT / 128), 256, 0, stream>>>(
      ff1, w2_t, BT, D, F, partial, (float*)d_out, nullptr);
}
